// MultiHeadSelfAttention_45526653337863
// MI455X (gfx1250) — matrix operand order repaired and checked by reading, not compiled, not run on hardware
//
#include <hip/hip_runtime.h>
#include <hip/hip_bf16.h>

// ---- problem constants (from reference) ----
#define BB   4
#define SS   2048
#define DD   1024
#define HH   16
#define DKK  64

typedef __attribute__((ext_vector_type(16))) __bf16 v16bf;
typedef __attribute__((ext_vector_type(8)))  __bf16 v8bf;
typedef __attribute__((ext_vector_type(8)))  float  v8f;

#define NEG_INF (-__builtin_huge_valf())

#if defined(__AMDGCN__)
#define USE_ASYNC_LDS 1
#else
#define USE_ASYNC_LDS 0
#endif

__device__ __forceinline__ __bf16 f2bf(float f) { return (__bf16)f; }

#if USE_ASYNC_LDS
// GLOBAL_LOAD_ASYNC_TO_LDS_B128: VDST = LDS byte address (low 32 bits of the
// generic pointer = LDS offset per the flat->LDS aperture mapping), VADDR =
// 64-bit global address, SADDR = NULL ("off"). Tracked by ASYNCcnt.
__device__ __forceinline__ void async_b128(const __bf16* g, __bf16* l) {
    const unsigned loff            = (unsigned)(unsigned long long)l;
    const unsigned long long gaddr = (unsigned long long)g;
    asm volatile("global_load_async_to_lds_b128 %0, %1, off"
                 :: "v"(loff), "v"(gaddr) : "memory");
}
__device__ __forceinline__ void wait_async0() {
#if __has_builtin(__builtin_amdgcn_s_wait_asynccnt)
    __builtin_amdgcn_s_wait_asynccnt(0);
#else
    asm volatile("s_wait_asynccnt 0x0" ::: "memory");
#endif
}
#endif

// ---------------------------------------------------------------------------
// elementwise f32 -> bf16
// ---------------------------------------------------------------------------
__global__ void k_cvt_bf16(const float* __restrict__ x, __bf16* __restrict__ y, int n) {
    int i = blockIdx.x * blockDim.x + threadIdx.x;
    if (i < n) y[i] = f2bf(x[i]);
}

// ---------------------------------------------------------------------------
// W [K=1024, N=1024] f32 row-major  ->  WT [N,K] bf16 (transposed, converted)
// ---------------------------------------------------------------------------
__global__ void k_transp_bf16(const float* __restrict__ W, __bf16* __restrict__ WT) {
    __shared__ float t[16][17];
    const int k0 = blockIdx.y * 16, n0 = blockIdx.x * 16;
    t[threadIdx.y][threadIdx.x] = W[(size_t)(k0 + threadIdx.y) * DD + n0 + threadIdx.x];
    __syncthreads();
    WT[(size_t)(n0 + threadIdx.y) * DD + k0 + threadIdx.x] = f2bf(t[threadIdx.x][threadIdx.y]);
}

// ---------------------------------------------------------------------------
// zero-length flags: zl[b] = all(key_padding_mask[b,:])
// ---------------------------------------------------------------------------
__global__ void k_zerolen(const unsigned char* __restrict__ kpm, int* __restrict__ zl) {
    __shared__ int any_unmasked;
    const int b = blockIdx.x;
    if (threadIdx.x == 0) any_unmasked = 0;
    __syncthreads();
    for (int i = threadIdx.x; i < SS; i += blockDim.x)
        if (kpm[(size_t)b * SS + i] == 0) any_unmasked = 1;   // benign race
    __syncthreads();
    if (threadIdx.x == 0) zl[b] = !any_unmasked;
}

// ---------------------------------------------------------------------------
// bf16 WMMA GEMM:  C[M,N] = A[M,K] * W[K,N] + bias   (W given transposed: WT[N,K])
// block = 128 threads (4 waves), tile 64x64, wave -> 16 rows x 64 cols.
// B tile staged global->LDS with double-buffered async copies (ASYNCcnt).
// mode 0: outF[M,N] f32 row-major
// mode 1: outH bf16 -> [B,H,S,DK]  (M = b*S+s, N = h*DK+dk)
// mode 2: outH bf16 -> [B,H,DK,S]  (head-transposed, for V)
// ---------------------------------------------------------------------------
__global__ void k_gemm(const __bf16* __restrict__ A, const __bf16* __restrict__ WT,
                       const float* __restrict__ bias,
                       float* __restrict__ outF, __bf16* __restrict__ outH,
                       int Ktot, int Ntot, int mode) {
    const int tid  = threadIdx.x;
    const int lane = tid & 31;
    const int w    = tid >> 5;
    const int half = lane >> 4;      // 0 or 1
    const int lr   = lane & 15;
    const int m0   = blockIdx.y * 64 + w * 16;
    const int n0   = blockIdx.x * 64;

#if USE_ASYNC_LDS
    // btile[buf][col*32 + k], 64 cols x 32 k of bf16 = 4 KB per buffer
    __shared__ __bf16 btile[2][64 * 32];
#endif

    v8f acc[4] = {};
    const __bf16* arow = A + (size_t)(m0 + lr) * Ktot;

#if USE_ASYNC_LDS
    // issue async prefetch of B tile for k-range [kk, kk+32) into buffer buf
    auto prefetch = [&](int kk, int buf) {
#pragma unroll
        for (int c = 0; c < 2; ++c) {
            const int chunk = tid + c * 128;        // 0..255, 16B each
            const int col   = chunk >> 2;           // 0..63
            const int ko    = (chunk & 3) * 8;      // 0,8,16,24
            async_b128(WT + (size_t)(n0 + col) * Ktot + kk + ko,
                       &btile[buf][col * 32 + ko]);
        }
    };
    int cur = 0;
    prefetch(0, 0);
    wait_async0();
    __syncthreads();
#endif

    for (int kk = 0; kk < Ktot; kk += 32) {
#if USE_ASYNC_LDS
        if (kk + 32 < Ktot) prefetch(kk + 32, cur ^ 1);
#endif
        // A fragment: a[i]=A[row][kk + half*16 + i]
        v8bf lo = *(const v8bf*)(arow + kk + half * 16);
        v8bf hi = *(const v8bf*)(arow + kk + half * 16 + 8);
        v16bf a;
#pragma unroll
        for (int i = 0; i < 8; ++i) { a[i] = lo[i]; a[8 + i] = hi[i]; }
#pragma unroll
        for (int nc = 0; nc < 4; ++nc) {
            // B fragment: b[i] = W[kk + half*16 + i][col]
#if USE_ASYNC_LDS
            const v16bf b = *(const v16bf*)(&btile[cur][(nc * 16 + lr) * 32 + half * 16]);
#else
            const v16bf b = *(const v16bf*)(WT + (size_t)(n0 + nc * 16 + lr) * Ktot + kk + half * 16);
#endif
            acc[nc] = __builtin_amdgcn_wmma_f32_16x16x32_bf16(false, a, false, b,
                                                              (short)0, acc[nc], false, false);
        }
#if USE_ASYNC_LDS
        wait_async0();
        __syncthreads();
        cur ^= 1;
#endif
    }

#pragma unroll
    for (int nc = 0; nc < 4; ++nc) {
        const int col = n0 + nc * 16 + lr;
        const float bv = bias[col];
#pragma unroll
        for (int r = 0; r < 8; ++r) {
            const int row = m0 + r + half * 8;
            const float v = acc[nc][r] + bv;
            if (mode == 0) {
                outF[(size_t)row * Ntot + col] = v;
            } else {
                const int bb = row >> 11, s = row & (SS - 1);
                const int h  = col >> 6,  dk = col & (DKK - 1);
                if (mode == 1)
                    outH[(((size_t)(bb * HH + h)) * SS + s) * DKK + dk] = f2bf(v);
                else // mode 2: head-transposed [B,H,DK,S]
                    outH[(((size_t)(bb * HH + h)) * DKK + dk) * SS + s] = f2bf(v);
            }
        }
    }
}

// ---------------------------------------------------------------------------
// scores: per (bh, 16 query rows): raw masked scores -> attn buffer (f32),
// per-lane online softmax, single cross-lane merge at the end -> stats.
// 1 wave per block.
// ---------------------------------------------------------------------------
__global__ void k_scores(const __bf16* __restrict__ Q, const __bf16* __restrict__ K,
                         const unsigned char* __restrict__ amask,
                         const unsigned char* __restrict__ kpm,
                         const int* __restrict__ zl,
                         float* __restrict__ attn, float* __restrict__ stats) {
    const int lane = threadIdx.x & 31;
    const int half = lane >> 4;
    const int lr   = lane & 15;
    const int bh   = blockIdx.y;
    const int b    = bh >> 4;
    const int q0   = blockIdx.x * 16;

    const __bf16* qp = Q + (size_t)bh * SS * DKK;
    const __bf16* kp = K + (size_t)bh * SS * DKK;
    const int zero = zl[b];

    // A fragments (Q rows) for the two K-dim steps (dk 0..31, 32..63)
    v16bf aq[2];
#pragma unroll
    for (int ks = 0; ks < 2; ++ks) {
        v8bf lo = *(const v8bf*)(qp + (size_t)(q0 + lr) * DKK + ks * 32 + half * 16);
        v8bf hi = *(const v8bf*)(qp + (size_t)(q0 + lr) * DKK + ks * 32 + half * 16 + 8);
#pragma unroll
        for (int i = 0; i < 8; ++i) { aq[ks][i] = lo[i]; aq[ks][8 + i] = hi[i]; }
    }

    float m[8], l[8];
#pragma unroll
    for (int r = 0; r < 8; ++r) { m[r] = NEG_INF; l[r] = 0.f; }

    for (int n0 = 0; n0 < SS; n0 += 16) {
        v8f acc = {};
#pragma unroll
        for (int ks = 0; ks < 2; ++ks) {
            // b[i] = K[key][ks*32 + half*16 + i]
            const v16bf bfrag = *(const v16bf*)(kp + (size_t)(n0 + lr) * DKK + ks * 32 + half * 16);
            acc = __builtin_amdgcn_wmma_f32_16x16x32_bf16(false, aq[ks], false, bfrag,
                                                          (short)0, acc, false, false);
        }
        const int key = n0 + lr;
        const int km  = kpm[(size_t)b * SS + key];
#pragma unroll
        for (int r = 0; r < 8; ++r) {
            const int qrow = q0 + r + half * 8;
            float s = acc[r] * 0.125f;                       // 1/sqrt(DK)
            if (amask[(size_t)qrow * SS + key] | km) s = NEG_INF;
            if (zero) s = 0.f;
            attn[((size_t)bh * SS + qrow) * SS + key] = s;   // raw masked score

            // per-lane online update (this lane's subset of keys)
            const float mn = fmaxf(m[r], s);
            const float e  = (s == NEG_INF) ? 0.f : __expf(s - mn);
            const float sc = (m[r] == mn) ? 1.f : __expf(m[r] - mn);
            l[r] = l[r] * sc + e;
            m[r] = mn;
        }
    }

    // merge (m,l) across the 16 lanes of each half-wave
#pragma unroll
    for (int r = 0; r < 8; ++r) {
        float mm = m[r], ll = l[r];
#pragma unroll
        for (int d = 1; d < 16; d <<= 1) {
            const float mo = __shfl_xor(mm, d);
            const float lo = __shfl_xor(ll, d);
            const float mn = fmaxf(mm, mo);
            const float e1 = (mm == mn) ? 1.f : __expf(mm - mn);
            const float e2 = (mo == mn) ? 1.f : __expf(mo - mn);
            ll = ll * e1 + lo * e2;
            mm = mn;
        }
        m[r] = mm; l[r] = ll;
    }

    if (lr == 0) {
#pragma unroll
        for (int r = 0; r < 8; ++r) {
            const size_t idx = (size_t)bh * SS + q0 + r + half * 8;
            stats[idx * 2]     = m[r];
            stats[idx * 2 + 1] = l[r];
        }
    }
}

// ---------------------------------------------------------------------------
// context: normalize scores -> final attn probs (in place, f32), and
// ctx[q,dk] = P * V via bf16 WMMA.  V supplied head-transposed [B,H,DK,S]
// so every B fragment is one contiguous 32B load.
// ---------------------------------------------------------------------------
__global__ void k_ctx(const float* __restrict__ stats, const __bf16* __restrict__ VT,
                      float* __restrict__ attn, __bf16* __restrict__ ctxB) {
    const int lane = threadIdx.x & 31;
    const int half = lane >> 4;
    const int lr   = lane & 15;
    const int bh   = blockIdx.y;
    const int b    = bh >> 4;
    const int h    = bh & (HH - 1);
    const int q0   = blockIdx.x * 16;

    const __bf16* vtp = VT + (size_t)bh * SS * DKK;    // [DK][S]

    const size_t ridx = (size_t)bh * SS + q0 + lr;     // this lane's A-row
    const float mrow = stats[ridx * 2];
    const float lrow = stats[ridx * 2 + 1];
    const float rcp  = (lrow > 0.f) ? (1.f / lrow) : 0.f;
    const bool  ok   = (lrow > 0.f);

    float* srow = attn + ((size_t)bh * SS + q0 + lr) * SS;

    v8f acc[4] = {};
    for (int k0 = 0; k0 < SS; k0 += 32) {
        const int c0 = half * 16;
        float4 s0 = *(const float4*)(srow + k0 + c0);
        float4 s1 = *(const float4*)(srow + k0 + c0 + 4);
        float4 s2 = *(const float4*)(srow + k0 + c0 + 8);
        float4 s3 = *(const float4*)(srow + k0 + c0 + 12);
        float pv[16];
        const float* sp[4] = { &s0.x, &s1.x, &s2.x, &s3.x };
#pragma unroll
        for (int g = 0; g < 4; ++g)
#pragma unroll
            for (int i = 0; i < 4; ++i) {
                const float s = sp[g][i];
                pv[g * 4 + i] = (!ok || s == NEG_INF) ? 0.f : __expf(s - mrow) * rcp;
            }
        // write final probabilities (this is the returned attn matrix)
        *(float4*)(srow + k0 + c0)      = make_float4(pv[0], pv[1], pv[2], pv[3]);
        *(float4*)(srow + k0 + c0 + 4)  = make_float4(pv[4], pv[5], pv[6], pv[7]);
        *(float4*)(srow + k0 + c0 + 8)  = make_float4(pv[8], pv[9], pv[10], pv[11]);
        *(float4*)(srow + k0 + c0 + 12) = make_float4(pv[12], pv[13], pv[14], pv[15]);

        v16bf a;
#pragma unroll
        for (int i = 0; i < 16; ++i) a[i] = f2bf(pv[i]);

#pragma unroll
        for (int nc = 0; nc < 4; ++nc) {
            // b[i] = V[key = k0+half*16+i][dk = nc*16+lr] = VT[dk][key]  (contiguous)
            const v16bf bfrag = *(const v16bf*)(vtp + (size_t)(nc * 16 + lr) * SS + k0 + half * 16);
            acc[nc] = __builtin_amdgcn_wmma_f32_16x16x32_bf16(false, a, false, bfrag,
                                                              (short)0, acc[nc], false, false);
        }
    }

#pragma unroll
    for (int nc = 0; nc < 4; ++nc)
#pragma unroll
        for (int r = 0; r < 8; ++r) {
            const int row = q0 + r + half * 8;
            ctxB[((size_t)(b * SS + row)) * DD + h * DKK + nc * 16 + lr] = f2bf(acc[nc][r]);
        }
}

// ---------------------------------------------------------------------------
// launcher
// ---------------------------------------------------------------------------
extern "C" void kernel_launch(void* const* d_in, const int* in_sizes, int n_in,
                              void* d_out, int out_size, void* d_ws, size_t ws_size,
                              hipStream_t stream) {
    const float* src          = (const float*)d_in[0];
    const unsigned char* am   = (const unsigned char*)d_in[1];
    const unsigned char* kpm  = (const unsigned char*)d_in[2];
    const float* Wq = (const float*)d_in[3];  const float* bq = (const float*)d_in[4];
    const float* Wk = (const float*)d_in[5];  const float* bk = (const float*)d_in[6];
    const float* Wv = (const float*)d_in[7];  const float* bv = (const float*)d_in[8];
    const float* Wo = (const float*)d_in[9];  const float* bo = (const float*)d_in[10];

    float* out  = (float*)d_out;
    float* attn = out + (size_t)BB * SS * DD;          // tuple: (output, attn)

    // ---- workspace layout ----
    char* ws = (char*)d_ws;
    const size_t BSD   = (size_t)BB * SS * DD;         // 8,388,608 elems
    size_t off = 0;
    __bf16* srcB = (__bf16*)(ws + off); off += BSD * 2;
    __bf16* WqT  = (__bf16*)(ws + off); off += (size_t)DD * DD * 2;
    __bf16* WkT  = (__bf16*)(ws + off); off += (size_t)DD * DD * 2;
    __bf16* WvT  = (__bf16*)(ws + off); off += (size_t)DD * DD * 2;
    __bf16* WoT  = (__bf16*)(ws + off); off += (size_t)DD * DD * 2;
    __bf16* Qb   = (__bf16*)(ws + off); off += BSD * 2;
    __bf16* Kb   = (__bf16*)(ws + off); off += BSD * 2;
    __bf16* Vt   = (__bf16*)(ws + off); off += BSD * 2;      // [B,H,DK,S]
    float*  stats = (float*)(ws + off); off += (size_t)BB * HH * SS * 2 * 4;
    int*    zl    = (int*)(ws + off);   off += 256;
    __bf16* ctxB  = srcB;   // srcB dead after QKV GEMMs; reuse for context

    // 1) conversions
    k_cvt_bf16<<<dim3((unsigned)((BSD + 255) / 256)), dim3(256), 0, stream>>>(src, srcB, (int)BSD);
    dim3 tg(DD / 16, DD / 16), tb(16, 16);
    k_transp_bf16<<<tg, tb, 0, stream>>>(Wq, WqT);
    k_transp_bf16<<<tg, tb, 0, stream>>>(Wk, WkT);
    k_transp_bf16<<<tg, tb, 0, stream>>>(Wv, WvT);
    k_transp_bf16<<<tg, tb, 0, stream>>>(Wo, WoT);
    k_zerolen<<<dim3(BB), dim3(256), 0, stream>>>(kpm, zl);

    // 2) QKV projections: [8192,1024]x[1024,1024]
    dim3 gg(DD / 64, (BB * SS) / 64);   // (16, 128)
    k_gemm<<<gg, dim3(128), 0, stream>>>(srcB, WqT, bq, nullptr, Qb, DD, DD, 1);
    k_gemm<<<gg, dim3(128), 0, stream>>>(srcB, WkT, bk, nullptr, Kb, DD, DD, 1);
    k_gemm<<<gg, dim3(128), 0, stream>>>(srcB, WvT, bv, nullptr, Vt, DD, DD, 2); // V transposed

    // 3) scores + online softmax stats
    dim3 ag(SS / 16, BB * HH);          // (128, 64)
    k_scores<<<ag, dim3(32), 0, stream>>>(Qb, Kb, am, kpm, zl, attn, stats);

    // 4) normalize (final attn probs) + P*V context
    k_ctx<<<ag, dim3(32), 0, stream>>>(stats, Vt, attn, ctxB);

    // 5) output projection -> d_out (f32)
    k_gemm<<<gg, dim3(128), 0, stream>>>(ctxB, WoT, bo, out, nullptr, DD, DD, 0);
}
